// w_o_MHA_28063316312601
// MI455X (gfx1250) — matrix operand order repaired and checked by reading, not compiled, not run on hardware
//
#include <hip/hip_runtime.h>
#include <hip/hip_bf16.h>
#include <math.h>

// ---------------------------------------------------------------------------
// Types for WMMA fragments (gfx1250, wave32)
// ---------------------------------------------------------------------------
typedef __bf16 bf16_t;
typedef __attribute__((ext_vector_type(8)))  bf16_t v8bf;
typedef __attribute__((ext_vector_type(16))) bf16_t v16bf;
typedef __attribute__((ext_vector_type(8)))  float  v8f;

__device__ __forceinline__ unsigned short f2bf_bits(float f) {
    unsigned int u = __builtin_bit_cast(unsigned int, f);
    unsigned int r = u + 0x7FFFu + ((u >> 16) & 1u);   // round-to-nearest-even
    return (unsigned short)(r >> 16);
}

// ---------------------------------------------------------------------------
// Elementwise f32 -> bf16 (activations). n must be a multiple of 4.
// ---------------------------------------------------------------------------
__global__ __launch_bounds__(256)
void cvt_f32_bf16(const float* __restrict__ in, unsigned short* __restrict__ out,
                  long long n)
{
    long long i = ((long long)blockIdx.x * blockDim.x + threadIdx.x) * 4;
    if (i + 3 >= n) return;
    float4 v = *(const float4*)(in + i);
    out[i + 0] = f2bf_bits(v.x);
    out[i + 1] = f2bf_bits(v.y);
    out[i + 2] = f2bf_bits(v.z);
    out[i + 3] = f2bf_bits(v.w);
}

// ---------------------------------------------------------------------------
// Transpose + convert: in [E][K][N] f32 -> out [E][N][K] bf16.
// 32x32 LDS tile transpose; K, N multiples of 32.
// ---------------------------------------------------------------------------
__global__ __launch_bounds__(256)
void transpose_cvt(const float* __restrict__ in, unsigned short* __restrict__ out,
                   int K, int N)
{
    __shared__ float tile[32][33];
    const float* src = in + (size_t)blockIdx.z * K * N;
    unsigned short* dst = out + (size_t)blockIdx.z * N * K;
    int k0 = blockIdx.y * 32, n0 = blockIdx.x * 32;
    int tx = threadIdx.x & 31, ty = threadIdx.x >> 5;   // 32 x 8 threads
    #pragma unroll
    for (int i = 0; i < 32; i += 8)
        tile[ty + i][tx] = src[(size_t)(k0 + ty + i) * N + n0 + tx];
    __syncthreads();
    #pragma unroll
    for (int i = 0; i < 32; i += 8)
        dst[(size_t)(n0 + ty + i) * K + k0 + tx] = f2bf_bits(tile[tx][ty + i]);
}

// ---------------------------------------------------------------------------
// Tiled GEMM (all-bf16 operands): C = act(A[M,K] @ B[K,N] + bias[N])
//   A: bf16 [M][K] row-major.  Bt: bf16 [N][K] (pre-transposed weight).
//   Output: f32 to Cf and/or bf16 to Cb (either may be null).
// Block tile 128x128x32, 256 threads = 8 waves, wave tile 64(M) x 32(N).
// Double-buffered LDS; one barrier per k-step; global loads for step k+1
// issued before the WMMA burst for step k.
// Requires M%128==0, N%128==0, K%32==0 (true for every GEMM in this net).
// ---------------------------------------------------------------------------
#define BM 128
#define BN 128
#define BK 32

__global__ __launch_bounds__(256)
void gemm_bf16_wmma(const unsigned short* __restrict__ A,
                    const unsigned short* __restrict__ Bt,
                    const float* __restrict__ bias,
                    float* __restrict__ Cf, unsigned short* __restrict__ Cb,
                    int M, int N, int K, int relu)
{
    __shared__ unsigned short lsA[2][BM * BK];  // [m][k] bf16
    __shared__ unsigned short lsB[2][BN * BK];  // [n][k] bf16

    const int tid  = threadIdx.x;
    const int lane = tid & 31;
    const int wave = tid >> 5;     // 0..7
    const int wm   = wave & 1;     // 64-row half
    const int wn   = wave >> 1;    // 32-col quarter
    const int bm0  = blockIdx.y * BM;
    const int bn0  = blockIdx.x * BN;

    const int halfsel = (lane >> 4) & 1;
    const int l15     = lane & 15;

    // staging geometry: 512 16B chunks per 128x32 bf16 tile; 2 per thread
    const int srow = tid >> 2;         // 0..63
    const int scol = (tid & 3) * 8;    // bf16 element column of 16B chunk
    const unsigned short* pa = A  + (size_t)(bm0 + srow) * K + scol;
    const unsigned short* pb = Bt + (size_t)(bn0 + srow) * K + scol;
    const int lofs = srow * BK + scol;

    v8f acc[4][2] = {};
    uint4 ra0, ra1, rb0, rb1;

    // prologue: stage k-step 0 into buffer 0
    ra0 = *(const uint4*)(pa);
    ra1 = *(const uint4*)(pa + (size_t)64 * K);
    rb0 = *(const uint4*)(pb);
    rb1 = *(const uint4*)(pb + (size_t)64 * K);
    *(uint4*)&lsA[0][lofs]           = ra0;
    *(uint4*)&lsA[0][lofs + 64 * BK] = ra1;
    *(uint4*)&lsB[0][lofs]           = rb0;
    *(uint4*)&lsB[0][lofs + 64 * BK] = rb1;

    const int nk = K / BK;
    for (int kb = 0; kb < nk; ++kb) {
        const int buf = kb & 1;
        __syncthreads();   // staging of buf complete (and prior compute done)

        if (kb + 1 < nk) { // issue next tile's global loads before WMMA burst
            int ko = (kb + 1) * BK;
            ra0 = *(const uint4*)(pa + ko);
            ra1 = *(const uint4*)(pa + (size_t)64 * K + ko);
            rb0 = *(const uint4*)(pb + ko);
            rb1 = *(const uint4*)(pb + (size_t)64 * K + ko);
        }

        // ---- fragments per ISA 7.12.2 ----
        // A 16x32 bf16: lanes 0-15 -> M=l, K={0..7,16..23}; lanes 16-31 -> K={8..15,24..31}
        v16bf af[4];
        #pragma unroll
        for (int mi = 0; mi < 4; ++mi) {
            int row = wm * 64 + mi * 16 + l15;
            const v8bf* fp = (const v8bf*)&lsA[buf][row * BK + halfsel * 8];
            v8bf lo = fp[0];
            v8bf hi = fp[2];
            af[mi] = __builtin_shufflevector(lo, hi, 0,1,2,3,4,5,6,7,8,9,10,11,12,13,14,15);
        }
        // B 32x16 bf16: lanes 0-15 -> N=l, K={0..7,16..23}; lanes 16-31 -> K={8..15,24..31}
        v16bf bfr[2];
        #pragma unroll
        for (int ni = 0; ni < 2; ++ni) {
            int col = wn * 32 + ni * 16 + l15;
            const v8bf* fp = (const v8bf*)&lsB[buf][col * BK + halfsel * 8];
            v8bf lo = fp[0];
            v8bf hi = fp[2];
            bfr[ni] = __builtin_shufflevector(lo, hi, 0,1,2,3,4,5,6,7,8,9,10,11,12,13,14,15);
        }
        #pragma unroll
        for (int mi = 0; mi < 4; ++mi)
            #pragma unroll
            for (int ni = 0; ni < 2; ++ni)
                acc[mi][ni] = __builtin_amdgcn_wmma_f32_16x16x32_bf16(
                    false, af[mi], false, bfr[ni], (short)0, acc[mi][ni], false, false);

        if (kb + 1 < nk) { // store next tile into the other buffer
            int obuf = buf ^ 1;
            *(uint4*)&lsA[obuf][lofs]           = ra0;
            *(uint4*)&lsA[obuf][lofs + 64 * BK] = ra1;
            *(uint4*)&lsB[obuf][lofs]           = rb0;
            *(uint4*)&lsB[obuf][lofs + 64 * BK] = rb1;
        }
    }

    // ---- epilogue: C/D f32 16x16 layout ----
    #pragma unroll
    for (int mi = 0; mi < 4; ++mi) {
        #pragma unroll
        for (int ni = 0; ni < 2; ++ni) {
            int n = bn0 + wn * 32 + ni * 16 + l15;
            float bb = bias ? bias[n] : 0.0f;
            int mbase = bm0 + wm * 64 + mi * 16 + halfsel * 8;
            #pragma unroll
            for (int r = 0; r < 8; ++r) {
                float v = acc[mi][ni][r] + bb;
                if (relu) v = fmaxf(v, 0.0f);
                size_t o = (size_t)(mbase + r) * N + n;
                if (Cf) Cf[o] = v;
                if (Cb) Cb[o] = f2bf_bits(v);
            }
        }
    }
}

// ---------------------------------------------------------------------------
// Gate: G[b, 0..7] = softmax(X[b,:] @ W[:,8] + bias).  One wave per row.
// ---------------------------------------------------------------------------
__global__ __launch_bounds__(256)
void gate_softmax(const float* __restrict__ X, const float* __restrict__ W,
                  const float* __restrict__ bias, float* __restrict__ G,
                  int nrows, int Din)
{
    int wid  = (blockIdx.x * blockDim.x + threadIdx.x) >> 5;
    int lane = threadIdx.x & 31;
    if (wid >= nrows) return;
    const float* x = X + (size_t)wid * Din;
    float acc[8] = {0,0,0,0,0,0,0,0};
    for (int k = lane; k < Din; k += 32) {
        float xv = x[k];
        const float* wr = W + (size_t)k * 8;
        #pragma unroll
        for (int e = 0; e < 8; ++e) acc[e] += xv * wr[e];
    }
    #pragma unroll
    for (int s = 16; s >= 1; s >>= 1)
        #pragma unroll
        for (int e = 0; e < 8; ++e) acc[e] += __shfl_xor(acc[e], s, 32);
    if (lane == 0) {
        float mx = -1e30f;
        #pragma unroll
        for (int e = 0; e < 8; ++e) { acc[e] += bias[e]; mx = fmaxf(mx, acc[e]); }
        float sum = 0.0f;
        #pragma unroll
        for (int e = 0; e < 8; ++e) { acc[e] = __expf(acc[e] - mx); sum += acc[e]; }
        float inv = 1.0f / sum;
        #pragma unroll
        for (int e = 0; e < 8; ++e) G[(size_t)wid * 8 + e] = acc[e] * inv;
    }
}

// ---------------------------------------------------------------------------
// Gated combine: out[b,c] = sum_e sp[e][b,c]*g[b,e] + sum_e sh[e][b,c]*g[b,4+e]
// Writes f32 (always) and optionally bf16 (for the next level's GEMM A).
// ---------------------------------------------------------------------------
__global__ __launch_bounds__(256)
void moe_combine(const float* __restrict__ sp, const float* __restrict__ sh,
                 const float* __restrict__ G, float* __restrict__ out,
                 unsigned short* __restrict__ outb,
                 long long estride, int Brows, int C)
{
    long long idx = (long long)blockIdx.x * blockDim.x + threadIdx.x;
    if (idx >= (long long)Brows * C) return;
    int b = (int)(idx / C);
    int c = (int)(idx % C);
    const float* g = G + (size_t)b * 8;
    size_t base = (size_t)b * C + c;
    float v = 0.0f;
    #pragma unroll
    for (int e = 0; e < 4; ++e) v += sp[(size_t)e * estride + base] * g[e];
    #pragma unroll
    for (int e = 0; e < 4; ++e) v += sh[(size_t)e * estride + base] * g[4 + e];
    out[idx] = v;
    if (outb) outb[idx] = f2bf_bits(v);
}

// ---------------------------------------------------------------------------
// Tower: y = sigmoid(relu(x@W1+b1) @ W2 + b2); one thread per row.
// ---------------------------------------------------------------------------
__global__ __launch_bounds__(256)
void tower_head(const float* __restrict__ X, const float* __restrict__ W1,
                const float* __restrict__ b1, const float* __restrict__ W2,
                const float* __restrict__ b2, float* __restrict__ Y,
                int Brows, int C)
{
    int idx = blockIdx.x * blockDim.x + threadIdx.x;
    if (idx >= 2 * Brows) return;
    int t = idx / Brows;
    const float* x  = X  + (size_t)idx * C;
    const float* w1 = W1 + (size_t)t * C * 64;
    const float* w2 = W2 + (size_t)t * 64;
    float hacc[64];
    #pragma unroll
    for (int j = 0; j < 64; ++j) hacc[j] = b1[t * 64 + j];
    for (int k = 0; k < C; ++k) {
        float xv = x[k];
        const float* wr = w1 + (size_t)k * 64;
        #pragma unroll
        for (int j = 0; j < 64; ++j) hacc[j] += xv * wr[j];
    }
    float y = b2[t];
    #pragma unroll
    for (int j = 0; j < 64; ++j) y += fmaxf(hacc[j], 0.0f) * w2[j];
    Y[idx] = 1.0f / (1.0f + __expf(-y));
}

// ---------------------------------------------------------------------------
// Host-side orchestration
// ---------------------------------------------------------------------------
static inline void launch_gemm(const unsigned short* A, const unsigned short* Bt,
                               const float* bias, float* Cf, unsigned short* Cb,
                               int M, int N, int K, int relu, hipStream_t s)
{
    dim3 grid(N / BN, M / BM);
    gemm_bf16_wmma<<<grid, 256, 0, s>>>(A, Bt, bias, Cf, Cb, M, N, K, relu);
}

static inline void launch_transpose(const float* in, unsigned short* out,
                                    int E, int K, int N, hipStream_t s)
{
    dim3 grid(N / 32, K / 32, E);
    transpose_cvt<<<grid, 256, 0, s>>>(in, out, K, N);
}

extern "C" void kernel_launch(void* const* d_in, const int* in_sizes, int n_in,
                              void* d_out, int out_size, void* d_ws, size_t ws_size,
                              hipStream_t stream)
{
    (void)in_sizes; (void)n_in; (void)out_size; (void)ws_size;
    const int B = 8192, D = 512, H = 1024, C0 = 256, C1 = 128;
    const int Btot = 2 * B;

    const float* x0 = (const float*)d_in[0];
    const float* x1 = (const float*)d_in[1];
    const float* xs = (const float*)d_in[2];
    const float* l0_shW1 = (const float*)d_in[3];   // [4,512,1024]
    const float* l0_shb1 = (const float*)d_in[4];
    const float* l0_shW2 = (const float*)d_in[5];   // [4,1024,256]
    const float* l0_shb2 = (const float*)d_in[6];
    const float* l0_spW1 = (const float*)d_in[7];   // [2,4,512,1024]
    const float* l0_spb1 = (const float*)d_in[8];
    const float* l0_spW2 = (const float*)d_in[9];   // [2,4,1024,256]
    const float* l0_spb2 = (const float*)d_in[10];
    const float* l0_gW   = (const float*)d_in[11];  // [3,512,8]
    const float* l0_gb   = (const float*)d_in[12];
    const float* l1_shW1 = (const float*)d_in[13];  // [4,256,1024]
    const float* l1_shb1 = (const float*)d_in[14];
    const float* l1_shW2 = (const float*)d_in[15];  // [4,1024,128]
    const float* l1_shb2 = (const float*)d_in[16];
    const float* l1_spW1 = (const float*)d_in[17];  // [2,4,256,1024]
    const float* l1_spb1 = (const float*)d_in[18];
    const float* l1_spW2 = (const float*)d_in[19];  // [2,4,1024,128]
    const float* l1_spb2 = (const float*)d_in[20];
    const float* l1_gW   = (const float*)d_in[21];  // [2,256,8]
    const float* l1_gb   = (const float*)d_in[22];
    const float* twW1 = (const float*)d_in[23];     // [2,128,64]
    const float* twb1 = (const float*)d_in[24];
    const float* twW2 = (const float*)d_in[25];     // [2,64,1]
    const float* twb2 = (const float*)d_in[26];

    float* y = (float*)d_out;

    // byte-granular bump allocator, 16B aligned
    char* wsb = (char*)d_ws;
    size_t off = 0;
    auto allocB = [&](size_t bytes) {
        char* p = wsb + off;
        off += (bytes + 15) & ~(size_t)15;
        return p;
    };
    auto allocF = [&](size_t n) { return (float*)allocB(n * 4); };
    auto allocH = [&](size_t n) { return (unsigned short*)allocB(n * 2); };

    // f32 buffers
    float* sho    = allocF((size_t)4 * Btot * C0);   // shared expert outs (reused lvl1)
    float* spo    = allocF((size_t)4 * Btot * C0);   // specific expert outs [e][t*B+b][c]
    float* g_t0   = allocF((size_t)B * 8);
    float* g_t1   = allocF((size_t)B * 8);
    float* g_sh   = allocF((size_t)Btot * 8);
    float* data0  = allocF((size_t)B * C0);
    float* data1  = allocF((size_t)B * C0);
    float* datash = allocF((size_t)Btot * C0);
    float* out1   = allocF((size_t)2 * B * C1);
    // bf16 buffers
    unsigned short* xs_bf   = allocH((size_t)Btot * D);
    unsigned short* h_bf    = allocH((size_t)Btot * H);     // fc1 output (reused)
    unsigned short* data0b  = allocH((size_t)B * C0);
    unsigned short* data1b  = allocH((size_t)B * C0);
    unsigned short* datashb = allocH((size_t)Btot * C0);
    unsigned short* w1t_sh0 = allocH((size_t)4 * H * D);    // [e][H][D]
    unsigned short* w1t_sp0 = allocH((size_t)8 * H * D);
    unsigned short* w2t_sh0 = allocH((size_t)4 * C0 * H);
    unsigned short* w2t_sp0 = allocH((size_t)8 * C0 * H);
    unsigned short* w1t_sh1 = allocH((size_t)4 * H * C0);
    unsigned short* w1t_sp1 = allocH((size_t)8 * H * C0);
    unsigned short* w2t_sh1 = allocH((size_t)4 * C1 * H);
    unsigned short* w2t_sp1 = allocH((size_t)8 * C1 * H);

    // ---- one-time conversions (cheap vs GEMM traffic) ----
    {
        long long n = (long long)Btot * D;
        cvt_f32_bf16<<<(int)(n / (4 * 256)), 256, 0, stream>>>(xs, xs_bf, n);
    }
    launch_transpose(l0_shW1, w1t_sh0, 4, D, H, stream);
    launch_transpose(l0_spW1, w1t_sp0, 8, D, H, stream);
    launch_transpose(l0_shW2, w2t_sh0, 4, H, C0, stream);
    launch_transpose(l0_spW2, w2t_sp0, 8, H, C0, stream);
    launch_transpose(l1_shW1, w1t_sh1, 4, C0, H, stream);
    launch_transpose(l1_spW1, w1t_sp1, 8, C0, H, stream);
    launch_transpose(l1_shW2, w2t_sh1, 4, H, C1, stream);
    launch_transpose(l1_spW2, w2t_sp1, 8, H, C1, stream);

    // ================= level 0 =================
    for (int e = 0; e < 4; ++e) {   // shared experts on xs [Btot, D]
        launch_gemm(xs_bf, w1t_sh0 + (size_t)e * H * D, l0_shb1 + (size_t)e * H,
                    nullptr, h_bf, Btot, H, D, 1, stream);
        launch_gemm(h_bf, w2t_sh0 + (size_t)e * C0 * H, l0_shb2 + (size_t)e * C0,
                    sho + (size_t)e * Btot * C0, nullptr, Btot, C0, H, 0, stream);
    }
    for (int t = 0; t < 2; ++t) {   // specific experts: x_t = rows t*B.. of xs
        const unsigned short* xt = xs_bf + (size_t)t * B * D;
        for (int e = 0; e < 4; ++e) {
            size_t we = (size_t)(t * 4 + e);
            launch_gemm(xt, w1t_sp0 + we * H * D, l0_spb1 + we * H,
                        nullptr, h_bf, B, H, D, 1, stream);
            launch_gemm(h_bf, w2t_sp0 + we * C0 * H, l0_spb2 + we * C0,
                        spo + (size_t)e * Btot * C0 + (size_t)t * B * C0, nullptr,
                        B, C0, H, 0, stream);
        }
    }
    gate_softmax<<<B / 8, 256, 0, stream>>>(x0, l0_gW + 0 * (size_t)D * 8, l0_gb + 0,  g_t0, B, D);
    gate_softmax<<<B / 8, 256, 0, stream>>>(x1, l0_gW + 1 * (size_t)D * 8, l0_gb + 8,  g_t1, B, D);
    gate_softmax<<<Btot / 8, 256, 0, stream>>>(xs, l0_gW + 2 * (size_t)D * 8, l0_gb + 16, g_sh, Btot, D);
    {
        long long es = (long long)Btot * C0;
        moe_combine<<<(int)((long long)B * C0 / 256), 256, 0, stream>>>(
            spo + 0 * (size_t)B * C0, sho + 0 * (size_t)B * C0, g_t0, data0, data0b, es, B, C0);
        moe_combine<<<(int)((long long)B * C0 / 256), 256, 0, stream>>>(
            spo + 1 * (size_t)B * C0, sho + 1 * (size_t)B * C0, g_t1, data1, data1b, es, B, C0);
        moe_combine<<<(int)((long long)Btot * C0 / 256), 256, 0, stream>>>(
            spo, sho, g_sh, datash, datashb, es, Btot, C0);
    }

    // ================= level 1 (last: no shared gate) =================
    for (int e = 0; e < 4; ++e) {
        launch_gemm(datashb, w1t_sh1 + (size_t)e * H * C0, l1_shb1 + (size_t)e * H,
                    nullptr, h_bf, Btot, H, C0, 1, stream);
        launch_gemm(h_bf, w2t_sh1 + (size_t)e * C1 * H, l1_shb2 + (size_t)e * C1,
                    sho + (size_t)e * Btot * C1, nullptr, Btot, C1, H, 0, stream);
    }
    for (int t = 0; t < 2; ++t) {
        const unsigned short* dt = (t == 0) ? data0b : data1b;
        for (int e = 0; e < 4; ++e) {
            size_t we = (size_t)(t * 4 + e);
            launch_gemm(dt, w1t_sp1 + we * H * C0, l1_spb1 + we * H,
                        nullptr, h_bf, B, H, C0, 1, stream);
            launch_gemm(h_bf, w2t_sp1 + we * C1 * H, l1_spb2 + we * C1,
                        spo + (size_t)e * Btot * C1 + (size_t)t * B * C1, nullptr,
                        B, C1, H, 0, stream);
        }
    }
    gate_softmax<<<B / 8, 256, 0, stream>>>(data0, l1_gW + 0 * (size_t)C0 * 8, l1_gb + 0, g_t0, B, C0);
    gate_softmax<<<B / 8, 256, 0, stream>>>(data1, l1_gW + 1 * (size_t)C0 * 8, l1_gb + 8, g_t1, B, C0);
    {
        long long es = (long long)Btot * C1;
        moe_combine<<<(int)((long long)B * C1 / 256), 256, 0, stream>>>(
            spo + 0 * (size_t)B * C1, sho + 0 * (size_t)B * C1, g_t0,
            out1 + 0, nullptr, es, B, C1);
        moe_combine<<<(int)((long long)B * C1 / 256), 256, 0, stream>>>(
            spo + 1 * (size_t)B * C1, sho + 1 * (size_t)B * C1, g_t1,
            out1 + (size_t)B * C1, nullptr, es, B, C1);
    }

    // ================= towers =================
    tower_head<<<(2 * B + 255) / 256, 256, 0, stream>>>(out1, twW1, twb1, twW2, twb2, y, B, C1);
}
